// XORChainModel_74483322847757
// MI455X (gfx1250) — 8-bit matrix operand order repaired and checked by reading, compiled, not run on hardware
//
#include <hip/hip_runtime.h>
#include <hip/hip_bf16.h>
#include <stddef.h>


static constexpr int S      = 4096;
static constexpr int B      = 8192;
static constexpr int CHUNK  = 64;
static constexpr int NCHUNK = S / CHUNK;

typedef __attribute__((ext_vector_type(4))) int   i4;
typedef __attribute__((ext_vector_type(4))) float f4;
typedef __attribute__((ext_vector_type(8))) int   v8i;

#if __has_builtin(__builtin_amdgcn_global_load_async_to_lds_b128) && \
    __has_builtin(__builtin_amdgcn_s_wait_asynccnt)
#define HAVE_ASYNC 1
#else
#define HAVE_ASYNC 0
#endif

#if HAVE_ASYNC
typedef __attribute__((address_space(1))) i4 as1_i4;
typedef __attribute__((address_space(3))) i4 as3_i4;

__device__ __forceinline__ void issue_group(const int* gsrc, i4* lbase, int tid) {
#pragma unroll
  for (int g = 0; g < 8; ++g) {
    __builtin_amdgcn_global_load_async_to_lds_b128(
        (as1_i4*)(gsrc + (size_t)g * B),
        (as3_i4*)(lbase + g * 256 + tid),
        0, 0);
  }
}
#endif

__global__ __launch_bounds__(256) void xor_chunk_totals(const int* __restrict__ in,
                                                        int* __restrict__ ws) {
  const int tid     = threadIdx.x;
  const int col4    = (blockIdx.x * 256 + tid) * 4;
  const int rowbase = blockIdx.y * CHUNK;

  i4 acc = {0, 0, 0, 0};
#pragma unroll 8
  for (int r = 0; r < CHUNK; ++r) {
    const int* p = in + (size_t)(rowbase + r) * B + col4;
    __builtin_prefetch(in + (size_t)(rowbase + r + 16) * B + col4, 0, 0);
    acc ^= __builtin_nontemporal_load((const i4*)p);
  }
  *(i4*)(ws + (size_t)blockIdx.y * B + col4) = acc;
}

__global__ __launch_bounds__(256) void xor_scan_totals_wmma(int* __restrict__ ws) {
  const int lane    = threadIdx.x & 31;
  const int wid     = (blockIdx.x * 256 + threadIdx.x) >> 5;
  const int colbase = wid * 16;
  const int n       = lane & 15;
  const int hi      = lane >> 4;

  v8i b;
#pragma unroll
  for (int v = 0; v < 8; ++v) {
    int word     = 0;
    const int kb = ((v >= 4) ? 32 : 0) + hi * 16 + (v & 3) * 4;
#pragma unroll
    for (int by = 0; by < 4; ++by) {
      const int k = kb + by;
      word |= (ws[k * B + colbase + n] & 1) << (8 * by);
    }
    b[v] = word;
  }

  const v8i zero = {0, 0, 0, 0, 0, 0, 0, 0};

#pragma unroll
  for (int r = 0; r < 4; ++r) {
    const int thresh = r * 16 + n;
    v8i a;
#pragma unroll
    for (int v = 0; v < 8; ++v) {
      int word = 0;
#pragma unroll
      for (int by = 0; by < 4; ++by) {
        const int k = ((v >= 4) ? 32 : 0) + hi * 16 + (v & 3) * 4 + by;
        word |= (int)(k < thresh) << (8 * by);
      }
      a[v] = word;
    }

    v8i d = __builtin_amdgcn_wmma_i32_16x16x64_iu8(
         false, a,  false, b, zero,
         false,  false);

#pragma unroll
    for (int g = 0; g < 8; ++g) {
      const int p = r * 16 + g + hi * 8;
      ws[p * B + colbase + n] = d[g] & 1;
    }
  }
}

__global__ __launch_bounds__(256) void xor_emit(const int* __restrict__ in,
                                                const int* __restrict__ ws,
                                                float* __restrict__ out) {
  const int tid     = threadIdx.x;
  const int col4    = (blockIdx.x * 256 + tid) * 4;
  const int chunk   = blockIdx.y;
  const int rowbase = chunk * CHUNK;

  i4 acc = *(const i4*)(ws + (size_t)chunk * B + col4);

#if HAVE_ASYNC
  __shared__ i4 stage[2][8 * 256];
  const int* gbase = in + (size_t)rowbase * B + col4;
  issue_group(gbase, &stage[0][0], tid);
#pragma unroll
  for (int s = 0; s < 8; ++s) {
    if (s < 7) {
      issue_group(gbase + (size_t)(s + 1) * 8 * B, &stage[(s + 1) & 1][0], tid);
      __builtin_amdgcn_s_wait_asynccnt(8);
    } else {
      __builtin_amdgcn_s_wait_asynccnt(0);
    }
    const i4* sb = &stage[s & 1][0];
#pragma unroll
    for (int g = 0; g < 8; ++g) {
      acc ^= sb[g * 256 + tid];
      const f4 fv = __builtin_convertvector(acc, f4);
      __builtin_nontemporal_store(
          fv, (f4*)(out + (size_t)(rowbase + s * 8 + g) * B + col4));
    }
  }
#else
  const i4* src = (const i4*)(in + (size_t)rowbase * B + col4);
#pragma unroll 8
  for (int r = 0; r < CHUNK; ++r) {
    acc ^= __builtin_nontemporal_load(src + (size_t)r * (B / 4));
    const f4 fv = __builtin_convertvector(acc, f4);
    __builtin_nontemporal_store(
        fv, (f4*)(out + (size_t)(rowbase + r) * B + col4));
  }
#endif
}

extern "C" void kernel_launch(void* const* d_in, const int* in_sizes, int n_in,
                              void* d_out, int out_size, void* d_ws, size_t ws_size,
                              hipStream_t stream) {
  (void)in_sizes; (void)n_in; (void)out_size; (void)ws_size;
  const int* in  = (const int*)d_in[0];
  float*     out = (float*)d_out;
  int*       ws  = (int*)d_ws;

  dim3 grid(B / 1024, NCHUNK);
  xor_chunk_totals<<<grid, 256, 0, stream>>>(in, ws);
  xor_scan_totals_wmma<<<(B / 16) / 8, 256, 0, stream>>>(ws);
  xor_emit<<<grid, 256, 0, stream>>>(in, ws, out);
}
